// GraphRelMultiheadAtt_46505905881124
// MI455X (gfx1250) — matrix operand order repaired and checked by reading, not compiled, not run on hardware
//
#include <hip/hip_runtime.h>
#include <hip/hip_bf16.h>

// ---------------------------------------------------------------------------
// GraphRelMultiheadAtt for MI455X (gfx1250, wave32, WMMA bf16)
// Outputs (flat in d_out): out[2,2048,1024] | new_mem[2,2048,3072] | score[2,2048,2048,16]
// ---------------------------------------------------------------------------

typedef __attribute__((ext_vector_type(16))) __bf16 v16bf;
typedef __attribute__((ext_vector_type(4)))  __bf16 v4bf;
typedef __attribute__((ext_vector_type(8)))  float  v8f;

union Frag {
    v16bf  v;
    float4 f4[2];
};

#define LDSS 40   // padded LDS row stride in bf16 elements (80B, 16B aligned)

#define SEQ     2048
#define HID     1024
#define NH      16
#define HD      64
#define NEGF    (-60000.0f)

// ---------------------------------------------------------------------------
// GEMM: C[M,N] = A[M,K] * B[K,N].  ABF=false: A is f32; ABF=true: A is bf16.
// block = 128 threads (4 waves), tile 64(M) x 128(N), K-step 32.
// Per wave / K-chunk: 1 A-frag reused across 8 N-tiles -> 8 v_wmma.
// ---------------------------------------------------------------------------
template<bool ABF>
__global__ __launch_bounds__(128)
void gemm_kernel(const float* __restrict__ A, const __bf16* __restrict__ Abf,
                 const float* __restrict__ B, float* __restrict__ C,
                 int K, int lda, int ldb, int ldc)
{
    __shared__ alignas(16) __bf16 Asl[64 * LDSS];
    __shared__ alignas(16) __bf16 Bsl[128 * LDSS];

    const int tid  = threadIdx.x;
    const int wave = tid >> 5;
    const int lane = tid & 31;
    const int half = lane >> 4;
    const int l16  = lane & 15;
    const int m0 = blockIdx.y * 64;
    const int n0 = blockIdx.x * 128;

    v8f acc[8] = {};

    for (int k0 = 0; k0 < K; k0 += 32) {
        // stage A (64 rows x 32 k): float4 load -> packed v4bf (8B) LDS store
#pragma unroll
        for (int j = 0; j < 4; ++j) {
            int i = tid + j * 128;
            int r = i >> 3, c4 = (i & 7) << 2;
            v4bf h;
            if (ABF) {
                h = *(const v4bf*)&Abf[(size_t)(m0 + r) * lda + k0 + c4];
            } else {
                float4 f = *(const float4*)&A[(size_t)(m0 + r) * lda + k0 + c4];
                h.x = (__bf16)f.x; h.y = (__bf16)f.y;
                h.z = (__bf16)f.z; h.w = (__bf16)f.w;
            }
            *(v4bf*)&Asl[r * LDSS + c4] = h;
        }
        // stage B (32 k x 128 n) transposed: Bt[n][k]
#pragma unroll
        for (int j = 0; j < 8; ++j) {
            int i = tid + j * 128;
            int kk = i >> 5, n4 = (i & 31) << 2;
            float4 f = *(const float4*)&B[(size_t)(k0 + kk) * ldb + n0 + n4];
            Bsl[(n4 + 0) * LDSS + kk] = (__bf16)f.x;
            Bsl[(n4 + 1) * LDSS + kk] = (__bf16)f.y;
            Bsl[(n4 + 2) * LDSS + kk] = (__bf16)f.z;
            Bsl[(n4 + 3) * LDSS + kk] = (__bf16)f.w;
        }
        if (k0 + 32 < K) {  // global_prefetch_b8 of next K-slab
            if (ABF) __builtin_prefetch(&Abf[(size_t)(m0 + (tid >> 1)) * lda + k0 + 32], 0, 0);
            else     __builtin_prefetch(&A  [(size_t)(m0 + (tid >> 1)) * lda + k0 + 32], 0, 0);
            __builtin_prefetch(&B[(size_t)(k0 + 32 + (tid >> 2)) * ldb + n0 + (tid & 3) * 32], 0, 0);
        }
        __syncthreads();

        Frag fa;
        const __bf16* ar = &Asl[(wave * 16 + l16) * LDSS];
        fa.f4[0] = *(const float4*)(ar + 16 * half);
        fa.f4[1] = *(const float4*)(ar + 16 * half + 8);
        Frag fb[8];
#pragma unroll
        for (int nt = 0; nt < 8; ++nt) {
            const __bf16* br = &Bsl[(nt * 16 + l16) * LDSS];
            fb[nt].f4[0] = *(const float4*)(br + 16 * half);
            fb[nt].f4[1] = *(const float4*)(br + 16 * half + 8);
        }
#pragma unroll
        for (int nt = 0; nt < 8; ++nt)
            acc[nt] = __builtin_amdgcn_wmma_f32_16x16x32_bf16(
                false, fa.v, false, fb[nt].v, (short)0, acc[nt], false, false);
        __syncthreads();
    }

#pragma unroll
    for (int nt = 0; nt < 8; ++nt)
#pragma unroll
        for (int r = 0; r < 8; ++r) {
            int m = m0 + wave * 16 + r + 8 * half;
            int n = n0 + nt * 16 + l16;
            C[(size_t)m * ldc + n] = acc[nt][r];
        }
}

// ---------------------------------------------------------------------------
// new_mem[..., 2048:3072] = kv
// ---------------------------------------------------------------------------
__global__ __launch_bounds__(256)
void copy_kv_kernel(const float* __restrict__ kv, float* __restrict__ new_mem)
{
#pragma unroll
    for (int j = 0; j < 4; ++j) {
        size_t idx = (size_t)blockIdx.x * 1024 + threadIdx.x + j * 256;
        size_t m = idx >> 10, h = idx & 1023;
        new_mem[m * 3072 + 2048 + h] = kv[idx];
    }
}

// ---------------------------------------------------------------------------
// score[b,q,k,n] = (Q_n K_n^T)/8, masked -> NEG_FILL.  One (b,head) per z.
// Tile: 64 q x 128 k per block.
// ---------------------------------------------------------------------------
__global__ __launch_bounds__(128)
void score_kernel(const float* __restrict__ Qm, const float* __restrict__ newmem,
                  const int* __restrict__ mask, float* __restrict__ score)
{
    __shared__ alignas(16) __bf16 Qs[64 * LDSS];
    __shared__ alignas(16) __bf16 Ks[128 * LDSS];

    const int tid = threadIdx.x, wave = tid >> 5, lane = tid & 31;
    const int half = lane >> 4, l16 = lane & 15;
    const int b = blockIdx.z >> 4, head = blockIdx.z & 15;
    const int q0 = blockIdx.y * 64, k0 = blockIdx.x * 128;

    v8f acc[8] = {};

    for (int kc = 0; kc < HD; kc += 32) {
        // stage Q (64 rows x 32 d)
#pragma unroll
        for (int j = 0; j < 4; ++j) {
            int i = tid + j * 128;
            int r = i >> 3, c4 = (i & 7) << 2;
            float4 f = *(const float4*)&Qm[(size_t)(b * SEQ + q0 + r) * HID + head * HD + kc + c4];
            v4bf h;
            h.x = (__bf16)f.x; h.y = (__bf16)f.y; h.z = (__bf16)f.z; h.w = (__bf16)f.w;
            *(v4bf*)&Qs[r * LDSS + c4] = h;
        }
        // stage K (128 kpos x 32 d) -- already Bt layout
#pragma unroll
        for (int j = 0; j < 8; ++j) {
            int i = tid + j * 128;
            int r = i >> 3, c4 = (i & 7) << 2;
            float4 f = *(const float4*)&newmem[(size_t)(b * SEQ + k0 + r) * 3072 + head * HD + kc + c4];
            v4bf h;
            h.x = (__bf16)f.x; h.y = (__bf16)f.y; h.z = (__bf16)f.z; h.w = (__bf16)f.w;
            *(v4bf*)&Ks[r * LDSS + c4] = h;
        }
        __syncthreads();
        Frag fa;
        const __bf16* ar = &Qs[(wave * 16 + l16) * LDSS];
        fa.f4[0] = *(const float4*)(ar + 16 * half);
        fa.f4[1] = *(const float4*)(ar + 16 * half + 8);
        Frag fb[8];
#pragma unroll
        for (int nt = 0; nt < 8; ++nt) {
            const __bf16* br = &Ks[(nt * 16 + l16) * LDSS];
            fb[nt].f4[0] = *(const float4*)(br + 16 * half);
            fb[nt].f4[1] = *(const float4*)(br + 16 * half + 8);
        }
#pragma unroll
        for (int nt = 0; nt < 8; ++nt)
            acc[nt] = __builtin_amdgcn_wmma_f32_16x16x32_bf16(
                false, fa.v, false, fb[nt].v, (short)0, acc[nt], false, false);
        __syncthreads();
    }

#pragma unroll
    for (int nt = 0; nt < 8; ++nt)
#pragma unroll
        for (int r = 0; r < 8; ++r) {
            int q = q0 + wave * 16 + r + 8 * half;
            int k = k0 + nt * 16 + l16;
            float s = acc[nt][r] * 0.125f;   // 1/sqrt(64)
            if (mask[(size_t)(b * SEQ + q) * SEQ + k]) s = NEGF;
            score[((size_t)(b * SEQ + q) * SEQ + k) * NH + head] = s;
        }
}

// ---------------------------------------------------------------------------
// rowmax[b,q,n] = max_k score
// ---------------------------------------------------------------------------
__global__ __launch_bounds__(256)
void rowmax_kernel(const float* __restrict__ score, float* __restrict__ rowmax)
{
    const int bq = blockIdx.x, t = threadIdx.x;
    const int head = t & 15, chunk = t >> 4;
    const float* base = score + (size_t)bq * SEQ * NH + head;
    float m = -3.4e38f;
    for (int k = chunk * 128; k < chunk * 128 + 128; ++k)
        m = fmaxf(m, base[(size_t)k * NH]);
    __shared__ float red[256];
    red[t] = m;
    __syncthreads();
    for (int st = 128; st >= 16; st >>= 1) {
        if (t < st) red[t] = fmaxf(red[t], red[t + st]);
        __syncthreads();
    }
    if (t < 16) rowmax[(size_t)bq * NH + t] = red[t];
}

// ---------------------------------------------------------------------------
// attended[b,q,n,d] = softmax(score) @ V  (online exp + row sum, WMMA for P.V)
// ---------------------------------------------------------------------------
__global__ __launch_bounds__(128)
void pv_kernel(const float* __restrict__ score, const float* __restrict__ rowmax,
               const float* __restrict__ newmem, float* __restrict__ attended)
{
    __shared__ alignas(16) __bf16 Ps[64 * LDSS];
    __shared__ alignas(16) __bf16 Vs[64 * LDSS];
    __shared__ float rsum[64];

    const int tid = threadIdx.x, wave = tid >> 5, lane = tid & 31;
    const int half = lane >> 4, l16 = lane & 15;
    const int b = blockIdx.z, head = blockIdx.y;
    const int q0 = blockIdx.x * 64;

    if (tid < 64) rsum[tid] = 0.0f;
    __syncthreads();

    v8f acc[4] = {};

    for (int k0 = 0; k0 < SEQ; k0 += 32) {
        // stage P = exp(s - rowmax), bf16; all 32 lanes of a wave share one row
#pragma unroll
        for (int j = 0; j < 16; ++j) {
            int i = tid + j * 128;
            int r = i >> 5, c = i & 31;
            int q = q0 + r;
            float mx = rowmax[(size_t)(b * SEQ + q) * NH + head];
            float s  = score[((size_t)(b * SEQ + q) * SEQ + (k0 + c)) * NH + head];
            float e  = __expf(s - mx);
            Ps[r * LDSS + c] = (__bf16)e;
            for (int off = 16; off; off >>= 1) e += __shfl_xor(e, off, 32);
            if (lane == 0) atomicAdd(&rsum[r], e);
        }
        // stage V transposed: Vt[d][k] -- float4 over d, 4 scalar LDS stores
#pragma unroll
        for (int j = 0; j < 4; ++j) {
            int i = tid + j * 128;
            int kk = i >> 4, d4 = (i & 15) << 2;
            float4 f = *(const float4*)&newmem[(size_t)(b * SEQ + k0 + kk) * 3072 + 1024 + head * HD + d4];
            Vs[(d4 + 0) * LDSS + kk] = (__bf16)f.x;
            Vs[(d4 + 1) * LDSS + kk] = (__bf16)f.y;
            Vs[(d4 + 2) * LDSS + kk] = (__bf16)f.z;
            Vs[(d4 + 3) * LDSS + kk] = (__bf16)f.w;
        }
        __syncthreads();
        Frag fa;
        const __bf16* ar = &Ps[(wave * 16 + l16) * LDSS];
        fa.f4[0] = *(const float4*)(ar + 16 * half);
        fa.f4[1] = *(const float4*)(ar + 16 * half + 8);
#pragma unroll
        for (int nt = 0; nt < 4; ++nt) {
            Frag fb;
            const __bf16* br = &Vs[(nt * 16 + l16) * LDSS];
            fb.f4[0] = *(const float4*)(br + 16 * half);
            fb.f4[1] = *(const float4*)(br + 16 * half + 8);
            acc[nt] = __builtin_amdgcn_wmma_f32_16x16x32_bf16(
                false, fa.v, false, fb.v, (short)0, acc[nt], false, false);
        }
        __syncthreads();
    }

#pragma unroll
    for (int nt = 0; nt < 4; ++nt)
#pragma unroll
        for (int r = 0; r < 8; ++r) {
            int qr = wave * 16 + r + 8 * half;
            int q  = q0 + qr;
            int d  = nt * 16 + l16;
            attended[(size_t)(b * SEQ + q) * HID + head * HD + d] = acc[nt][r] / rsum[qr];
        }
}

// ---------------------------------------------------------------------------
// T5 relative-distance weights (masked, unnormalized) + row sums
// ---------------------------------------------------------------------------
__global__ __launch_bounds__(256)
void graph_dist_kernel(const int* __restrict__ mask, __bf16* __restrict__ distw,
                       float* __restrict__ growsum)
{
    const int bq = blockIdx.x;
    const int q = bq & (SEQ - 1);
    const float C = 64.0f / logf(8.0f);   // (NUM_BUCKETS-64)/log(512/64)
    float s = 0.0f;
    for (int k = threadIdx.x; k < SEQ; k += 256) {
        int rp = (k > q) ? (k - q) : (q - k);
        int bucket;
        if (rp < 64) bucket = rp;
        else {
            int lg = 64 + (int)(logf((float)rp * (1.0f / 64.0f)) * C);
            bucket = lg < 127 ? lg : 127;
        }
        float d = 1.0f / (float)(bucket + 1);
        if (mask[(size_t)bq * SEQ + k]) d = 0.0f;
        distw[(size_t)bq * SEQ + k] = (__bf16)d;
        s += d;
    }
    __shared__ float red[256];
    red[threadIdx.x] = s;
    __syncthreads();
    for (int st = 128; st; st >>= 1) {
        if (threadIdx.x < st) red[threadIdx.x] += red[threadIdx.x + st];
        __syncthreads();
    }
    if (threadIdx.x == 0) growsum[bq] = red[0];
}

__global__ __launch_bounds__(256)
void graph_norm_kernel(__bf16* __restrict__ distw, const float* __restrict__ growsum)
{
#pragma unroll
    for (int j = 0; j < 4; ++j) {
        size_t i = (size_t)blockIdx.x * 1024 + threadIdx.x + j * 256;
        int bq = (int)(i >> 11);
        distw[i] = (__bf16)((float)distw[i] / growsum[bq]);
    }
}

// ---------------------------------------------------------------------------
// x = query + att_out + graph_out; LayerNorm -> out
// ---------------------------------------------------------------------------
__global__ __launch_bounds__(256)
void ln_kernel(const float* __restrict__ query, const float* __restrict__ att,
               const float* __restrict__ graph, const float* __restrict__ g,
               const float* __restrict__ bta, float* __restrict__ out)
{
    const int row = blockIdx.x, t = threadIdx.x;
    float x[4];
    float s = 0.0f;
    __shared__ float red[256];
#pragma unroll
    for (int j = 0; j < 4; ++j) {
        size_t idx = (size_t)row * HID + t + j * 256;
        x[j] = query[idx] + att[idx] + graph[idx];
        s += x[j];
    }
    red[t] = s;
    __syncthreads();
    for (int st = 128; st; st >>= 1) {
        if (t < st) red[t] += red[t + st];
        __syncthreads();
    }
    float mu = red[0] * (1.0f / HID);
    __syncthreads();
    float v = 0.0f;
#pragma unroll
    for (int j = 0; j < 4; ++j) { float d = x[j] - mu; v += d * d; }
    red[t] = v;
    __syncthreads();
    for (int st = 128; st; st >>= 1) {
        if (t < st) red[t] += red[t + st];
        __syncthreads();
    }
    float rs = rsqrtf(red[0] * (1.0f / HID) + 1e-5f);
#pragma unroll
    for (int j = 0; j < 4; ++j) {
        int col = t + j * 256;
        size_t idx = (size_t)row * HID + col;
        out[idx] = (x[j] - mu) * rs * g[col] + bta[col];
    }
}

// ---------------------------------------------------------------------------
extern "C" void kernel_launch(void* const* d_in, const int* in_sizes, int n_in,
                              void* d_out, int out_size, void* d_ws, size_t ws_size,
                              hipStream_t stream)
{
    (void)in_sizes; (void)n_in; (void)out_size; (void)ws_size;
    const float* q    = (const float*)d_in[0];
    const float* kv   = (const float*)d_in[1];
    const int*   mask = (const int*)d_in[2];
    const float* W_kv = (const float*)d_in[3];
    const float* W_q  = (const float*)d_in[4];
    const float* W_o  = (const float*)d_in[5];
    const float* W_go = (const float*)d_in[6];
    const float* ln_g = (const float*)d_in[7];
    const float* ln_b = (const float*)d_in[8];

    float* out_f   = (float*)d_out;
    float* out_ln  = out_f;                     // [2,2048,1024]
    float* new_mem = out_f + (size_t)4194304;   // [2,2048,3072]
    float* score   = out_f + (size_t)16777216;  // [2,2048,2048,16]

    const size_t SZ = 4194304;                  // 2*2048*1024
    float*  ws      = (float*)d_ws;
    float*  wsQ     = ws;                       // query projection
    float*  wsAtt   = ws + SZ;                  // att_out
    float*  wsGraph = ws + 2 * SZ;              // graph_out
    float*  wsAttd  = ws + 3 * SZ;              // attended
    float*  wsTmp   = ws + 4 * SZ;              // w @ kv
    float*  wsRmax  = ws + 5 * SZ;              // [2,2048,16]
    float*  wsRsum  = wsRmax + 65536;           // [2,2048]
    __bf16* wsDist  = (__bf16*)(wsRsum + 4096); // [2,2048,2048] bf16

    dim3 blk128(128), blk256(256);

    // 1) query = q @ W_q
    gemm_kernel<false><<<dim3(8, 64), blk128, 0, stream>>>(
        q, nullptr, W_q, wsQ, 1024, 1024, 1024, 1024);
    // 2) kv_proj = kv @ W_kv -> new_mem[..., 0:2048]
    gemm_kernel<false><<<dim3(16, 64), blk128, 0, stream>>>(
        kv, nullptr, W_kv, new_mem, 1024, 1024, 2048, 3072);
    // 3) new_mem[..., 2048:3072] = kv
    copy_kv_kernel<<<dim3(4096), blk256, 0, stream>>>(kv, new_mem);
    // 4) masked scaled scores (output tensor)
    score_kernel<<<dim3(16, 32, 32), blk128, 0, stream>>>(wsQ, new_mem, mask, score);
    // 5) per-(b,q,head) max
    rowmax_kernel<<<dim3(4096), blk256, 0, stream>>>(score, wsRmax);
    // 6) attended = softmax(score) @ V
    pv_kernel<<<dim3(32, 16, 2), blk128, 0, stream>>>(score, wsRmax, new_mem, wsAttd);
    // 7) att_out = attended @ W_o
    gemm_kernel<false><<<dim3(8, 64), blk128, 0, stream>>>(
        wsAttd, nullptr, W_o, wsAtt, 1024, 1024, 1024, 1024);
    // 8) graph weights (unnormalized) + row sums
    graph_dist_kernel<<<dim3(4096), blk256, 0, stream>>>(mask, wsDist, wsRsum);
    // 9) normalize in place
    graph_norm_kernel<<<dim3(8192), blk256, 0, stream>>>(wsDist, wsRsum);
    // 10) tmp = w @ kv (per batch; bf16 A path)
    for (int b = 0; b < 2; ++b)
        gemm_kernel<true><<<dim3(8, 32), blk128, 0, stream>>>(
            nullptr, wsDist + (size_t)b * 2048 * 2048, kv + (size_t)b * 2048 * 1024,
            wsTmp + (size_t)b * 2048 * 1024, 2048, 2048, 1024, 1024);
    // 11) graph_out = tmp @ W_go
    gemm_kernel<false><<<dim3(8, 64), blk128, 0, stream>>>(
        wsTmp, nullptr, W_go, wsGraph, 1024, 1024, 1024, 1024);
    // 12) residual + LayerNorm
    ln_kernel<<<dim3(4096), blk256, 0, stream>>>(wsQ, wsAtt, wsGraph, ln_g, ln_b, out_ln);
}
